// AttentionBlock_59957743452249
// MI455X (gfx1250) — compile-verified, matrix operand order checked by reading, not run on hardware
//
#include <hip/hip_runtime.h>

typedef __attribute__((ext_vector_type(16))) __bf16 v16bf;
typedef __attribute__((ext_vector_type(8)))  __bf16 v8bf;
typedef __attribute__((ext_vector_type(8)))  float  v8f;
typedef __attribute__((ext_vector_type(4)))  float  v4f;

constexpr int BATCH = 4;
constexpr int CIN   = 512;   // input channels C
constexpr int SEQ   = 2048;  // sequence length S
constexpr int NF    = 512;   // key/value size
constexpr int RCH   = 128;   // rows per softmax chunk
constexpr int NCH   = SEQ / RCH;   // 16 chunks

union BF16x16 { v16bf v; v8bf h[2]; };

// ---- CDNA5 async global->LDS copy (16B per lane), tracked by ASYNCcnt ----
__device__ __forceinline__ void async_copy16(void* lds_ptr, const void* gptr)
{
    unsigned loff = (unsigned)(size_t)lds_ptr;            // low 32 bits = LDS offset
    unsigned long long ga = (unsigned long long)(size_t)gptr;
    asm volatile("global_load_async_to_lds_b128 %0, %1, off"
                 :: "v"(loff), "v"(ga) : "memory");
}
__device__ __forceinline__ void wait_async0()
{
    asm volatile("s_wait_asynccnt 0" ::: "memory");
}

// ------------------------------------------------------------------
// Kernel 1: fused QKV projection.
// W tile (16 n x 512 c, 16KB) preloaded once; x tile double-buffered
// (one barrier per K-step) with packed b32 DS stores.
// Q,K stored row-major [S,NF] bf16 (Q pre-scaled by 1/sqrt(NF));
// V stored transposed [NF,S] bf16 so k4 needs no transpose.
// ------------------------------------------------------------------
__global__ __launch_bounds__(256)
void k_qkv(const float* __restrict__ x,
           const float* __restrict__ Wq, const float* __restrict__ bq,
           const float* __restrict__ Wk, const float* __restrict__ bk,
           const float* __restrict__ Wv, const float* __restrict__ bv,
           char* __restrict__ qb_, char* __restrict__ kb_, char* __restrict__ vt_)
{
    __shared__ __bf16 xa[2][128][40];   // [buf][s][c] rows 80B: 16B-aligned chunks, banks spread
    __shared__ __bf16 wl[16][520];      // [n][c] full-C W tile, row 1040B (16B-aligned, bank skew 4)

    const int tid  = threadIdx.x;
    const int wave = tid >> 5;
    const int lane = tid & 31;
    const int lrow = lane & 15;
    const int hf   = lane >> 4;

    const int sblk = blockIdx.x * 128;
    const int n0   = blockIdx.y * 16;
    const int z    = blockIdx.z;
    const int b    = z / 3;
    const int m    = z % 3;             // 0=Q 1=K 2=V

    const float* W    = (m == 0) ? Wq : (m == 1) ? Wk : Wv;
    const float* bias = (m == 0) ? bq : (m == 1) ? bk : bv;

    const size_t xbase = (size_t)b * CIN * SEQ;
    const int s0 = sblk + wave * 16;

    // stage one 128s x 32c x-chunk (f32 -> packed bf16 pair -> ds_store_b32)
    auto stage = [&](int kk, __bf16 (*buf)[40]) {
#pragma unroll
        for (int i = 0; i < 8; ++i) {
            int idx = tid + i * 256;          // 2048 = 128 s * 16 c-pairs
            int cp  = idx >> 7;               // 0..15
            int sl  = idx & 127;
            const size_t g = xbase + (size_t)(kk * 32 + 2 * cp) * SEQ + sblk + sl;
            union { __bf16 h[2]; unsigned u; } pk;
            pk.h[0] = (__bf16)x[g];
            pk.h[1] = (__bf16)x[g + SEQ];
            *(unsigned*)&buf[sl][2 * cp] = pk.u;
        }
    };

    // preload entire W tile once (reused by all 16 K-steps)
#pragma unroll 4
    for (int i = 0; i < 32; ++i) {
        int idx = tid + i * 256;              // 8192 = 512 c * 16 n
        int nl = idx & 15;
        int cl = idx >> 4;
        wl[nl][cl] = (__bf16)W[(size_t)cl * NF + n0 + nl];
    }
    stage(0, xa[0]);
    __syncthreads();

    v8f acc = {};
    for (int kk = 0; kk < CIN / 32; ++kk) {
        if (kk + 1 < CIN / 32) stage(kk + 1, xa[(kk + 1) & 1]);
        const __bf16 (*xb)[40] = xa[kk & 1];
        BF16x16 a, bb;
        a.h[0]  = *(const v8bf*)&xb[wave * 16 + lrow][hf * 8];
        a.h[1]  = *(const v8bf*)&xb[wave * 16 + lrow][16 + hf * 8];
        bb.h[0] = *(const v8bf*)&wl[lrow][kk * 32 + hf * 8];
        bb.h[1] = *(const v8bf*)&wl[lrow][kk * 32 + 16 + hf * 8];
        acc = __builtin_amdgcn_wmma_f32_16x16x32_bf16(false, a.v, false, bb.v,
                                                      (short)0, acc, false, false);
        __syncthreads();
    }

    const float bval  = bias[n0 + lrow];
    const float scale = (m == 0) ? 0.04419417382415922f : 1.0f;   // 1/sqrt(512) folded into Q

    if (m < 2) {
        __bf16* dst = (__bf16*)((m == 0) ? qb_ : kb_)
                    + (size_t)b * SEQ * NF + (n0 + lrow);
#pragma unroll
        for (int r = 0; r < 8; ++r) {
            int s = s0 + 8 * hf + r;
            dst[(size_t)s * NF] = (__bf16)((acc[r] + bval) * scale);
        }
    } else {
        __bf16* vt = (__bf16*)vt_ + (size_t)b * NF * SEQ
                   + (size_t)(n0 + lrow) * SEQ + s0 + 8 * hf;
        v8bf o;
#pragma unroll
        for (int r = 0; r < 8; ++r) o[r] = (__bf16)(acc[r] + bval);
        *(v8bf*)vt = o;
    }
}

// ------------------------------------------------------------------
// Kernel 2: scores = (Q/sqrt d) . K^T, causal mask -> fp32 workspace.
// Block-shared 16x512 K tile staged ONCE via async global->LDS DMA
// (XOR-swizzled columns: conflict-free 16-lane b128 reads).
// ------------------------------------------------------------------
__global__ __launch_bounds__(256)
void k_scores(const char* __restrict__ qb_, const char* __restrict__ kb_,
              float* __restrict__ scores)
{
    __shared__ __bf16 klds[16][512];    // 16KB, row stride 1024B, XOR swizzle on 16B chunks

    const int tid  = threadIdx.x;
    const int wave = tid >> 5;
    const int lane = tid & 31;
    const int lrow = lane & 15;
    const int hf   = lane >> 4;

    const int k0   = blockIdx.x * 16;
    const int qblk = blockIdx.y * 128;
    const int b    = blockIdx.z;

    if (k0 > qblk + 127) return;        // block fully above diagonal (uniform)

    // async-stage K tile: 1024 16B chunks over 256 lanes
    const __bf16* kbase = (const __bf16*)kb_ + (size_t)b * SEQ * NF + (size_t)k0 * NF;
#pragma unroll
    for (int i = 0; i < 4; ++i) {
        int idx = tid + i * 256;
        int row = idx >> 6;             // 0..15
        int cc  = idx & 63;             // 16B chunk within row
        async_copy16((char*)klds + row * 1024 + ((cc ^ row) << 4),
                     kbase + (size_t)row * NF + cc * 8);
    }
    wait_async0();
    __syncthreads();

    const int q0 = qblk + wave * 16;
    if (k0 > q0 + 15) return;           // wave-uniform: EXEC stays full for WMMA

    const __bf16* qrow = (const __bf16*)qb_ + (size_t)b * SEQ * NF + (size_t)(q0 + lrow) * NF;

    v8f acc = {};
    for (int kk = 0; kk < NF / 32; ++kk) {
        BF16x16 a, bb;
        a.h[0] = *(const v8bf*)(qrow + kk * 32 + hf * 8);
        a.h[1] = *(const v8bf*)(qrow + kk * 32 + 16 + hf * 8);
        bb.h[0] = *(const v8bf*)((const char*)klds + lrow * 1024
                                 + (((kk * 4 + hf) ^ lrow) << 4));
        bb.h[1] = *(const v8bf*)((const char*)klds + lrow * 1024
                                 + (((kk * 4 + hf + 2) ^ lrow) << 4));
        if (kk + 1 < NF / 32) __builtin_prefetch(qrow + (kk + 1) * 32, 0, 0);
        acc = __builtin_amdgcn_wmma_f32_16x16x32_bf16(false, a.v, false, bb.v,
                                                      (short)0, acc, false, false);
    }

    const int sk = k0 + lrow;
    float* scol = scores + (size_t)b * SEQ * SEQ + sk;
#pragma unroll
    for (int r = 0; r < 8; ++r) {
        int sq = q0 + 8 * hf + r;
        scol[(size_t)sq * SEQ] = (sk <= sq) ? acc[r] : -__builtin_inff();
    }
}

// ------------------------------------------------------------------
// Kernel 3a: per-(column, 128-row-chunk) partial (max, sumexp).
// 512 blocks, fully coalesced across adjacent key columns.
// ------------------------------------------------------------------
__global__ __launch_bounds__(256)
void k_sm_part(const float* __restrict__ scores,
               float* __restrict__ pm, float* __restrict__ pl)
{
    const int k  = blockIdx.x * 256 + threadIdx.x;
    const int rc = blockIdx.y;
    const int b  = blockIdx.z;
    const int q0 = rc * RCH;
    const size_t o = ((size_t)b * NCH + rc) * SEQ + k;

    if (q0 + RCH - 1 < k) { pm[o] = -__builtin_inff(); pl[o] = 0.f; return; }

    const float* sc = scores + (size_t)b * SEQ * SEQ + k;
    float mx = -__builtin_inff();
    for (int i = 0; i < RCH; ++i) {
        int q = q0 + i;
        if (q >= k) mx = fmaxf(mx, sc[(size_t)q * SEQ]);
    }
    float l = 0.f;
    if (mx != -__builtin_inff()) {
        for (int i = 0; i < RCH; ++i) {
            int q = q0 + i;
            if (q >= k) l += __expf(sc[(size_t)q * SEQ] - mx);
        }
    }
    pm[o] = mx;
    pl[o] = l;
}

// ------------------------------------------------------------------
// Kernel 3b: combine 16 chunk partials per column -> (colmax, 1/colsum).
// ------------------------------------------------------------------
__global__ __launch_bounds__(256)
void k_sm_comb(const float* __restrict__ pm, const float* __restrict__ pl,
               float* __restrict__ cm, float* __restrict__ ci)
{
    const int k = blockIdx.x * 256 + threadIdx.x;
    const int b = blockIdx.y;
    float M = -__builtin_inff();
#pragma unroll
    for (int c = 0; c < NCH; ++c)
        M = fmaxf(M, pm[((size_t)b * NCH + c) * SEQ + k]);
    float L = 0.f;
#pragma unroll
    for (int c = 0; c < NCH; ++c) {
        float m = pm[((size_t)b * NCH + c) * SEQ + k];
        if (m != -__builtin_inff())
            L += pl[((size_t)b * NCH + c) * SEQ + k] * __expf(m - M);
    }
    cm[(size_t)b * SEQ + k] = M;
    ci[(size_t)b * SEQ + k] = 1.f / L;
}

// ------------------------------------------------------------------
// Kernel 3c: write normalized probs as bf16 (zeros above diagonal so
// k4 can run dense WMMA tiles). 512 parallel coalesced blocks.
// ------------------------------------------------------------------
__global__ __launch_bounds__(256)
void k_sm_write(const float* __restrict__ scores,
                const float* __restrict__ cm, const float* __restrict__ ci,
                char* __restrict__ p_)
{
    const int k  = blockIdx.x * 256 + threadIdx.x;
    const int rc = blockIdx.y;
    const int b  = blockIdx.z;
    const int q0 = rc * RCH;

    const float* sc = scores + (size_t)b * SEQ * SEQ + k;
    __bf16* P = (__bf16*)p_ + (size_t)b * SEQ * SEQ + k;
    const float M   = cm[(size_t)b * SEQ + k];
    const float inv = ci[(size_t)b * SEQ + k];

    for (int i = 0; i < RCH; ++i) {
        int q = q0 + i;
        float pv = (q >= k) ? __expf(sc[(size_t)q * SEQ] - M) * inv : 0.f;
        P[(size_t)q * SEQ] = (__bf16)pv;
    }
}

// ------------------------------------------------------------------
// Kernel 4: activation = P . V. Block-shared V^T tile (16 x <=2048,
// 64KB) async-staged to LDS with XOR swizzle; P rows loaded direct.
// Causal: only k < qblk+128 staged/iterated. Output written directly
// into [B, CIN+NF, S] (8 consecutive s -> two b128 stores).
// ------------------------------------------------------------------
__global__ __launch_bounds__(256)
void k_pv(const char* __restrict__ p_, const char* __restrict__ vt_,
          float* __restrict__ out)
{
    __shared__ __bf16 vlds[16][2048];   // 64KB, row stride 4096B, XOR swizzle on 16B chunks

    const int tid  = threadIdx.x;
    const int wave = tid >> 5;
    const int lane = tid & 31;
    const int lrow = lane & 15;
    const int hf   = lane >> 4;

    const int v0   = blockIdx.x * 16;
    const int qblk = blockIdx.y * 128;
    const int b    = blockIdx.z;

    const int kneed = qblk + 128;       // max key index + 1 used by this block
    const int cpr   = kneed >> 3;       // 16B chunks per row (multiple of 16)
    const __bf16* vbase = (const __bf16*)vt_ + (size_t)b * NF * SEQ + (size_t)v0 * SEQ;

    for (int idx = tid; idx < 16 * cpr; idx += 256) {
        int row = idx / cpr;
        int cc  = idx % cpr;
        async_copy16((char*)vlds + row * 4096 + ((cc ^ row) << 4),
                     vbase + (size_t)row * SEQ + cc * 8);
    }
    wait_async0();
    __syncthreads();

    const int q0 = qblk + wave * 16;
    const __bf16* prow = (const __bf16*)p_ + (size_t)b * SEQ * SEQ + (size_t)(q0 + lrow) * SEQ;

    const int kiters = (q0 + 16 + 31) >> 5;   // only k <= q0+15 can be non-zero
    v8f acc = {};
    for (int kk = 0; kk < kiters; ++kk) {
        BF16x16 a, bb;
        a.h[0] = *(const v8bf*)(prow + kk * 32 + hf * 8);
        a.h[1] = *(const v8bf*)(prow + kk * 32 + 16 + hf * 8);
        bb.h[0] = *(const v8bf*)((const char*)vlds + lrow * 4096
                                 + (((kk * 4 + hf) ^ lrow) << 4));
        bb.h[1] = *(const v8bf*)((const char*)vlds + lrow * 4096
                                 + (((kk * 4 + hf + 2) ^ lrow) << 4));
        acc = __builtin_amdgcn_wmma_f32_16x16x32_bf16(false, a.v, false, bb.v,
                                                      (short)0, acc, false, false);
    }

    float* o = out + (size_t)b * (CIN + NF) * SEQ
                   + (size_t)(CIN + v0 + lrow) * SEQ + q0 + 8 * hf;
    v4f lo, hi;
#pragma unroll
    for (int r = 0; r < 4; ++r) { lo[r] = acc[r]; hi[r] = acc[r + 4]; }
    *(v4f*)o       = lo;
    *(v4f*)(o + 4) = hi;
}

// ------------------------------------------------------------------
// Kernel 5: out[:, 0:CIN, :] = x  (float4 copy)
// ------------------------------------------------------------------
__global__ __launch_bounds__(256)
void k_copy_x(const float* __restrict__ x, float* __restrict__ out)
{
    const size_t i4  = (size_t)blockIdx.x * 256 + threadIdx.x;
    const size_t per = (size_t)CIN * SEQ / 4;
    const size_t b   = i4 / per;
    const size_t r   = i4 % per;
    const v4f* src = (const v4f*)x + b * per + r;
    v4f* dst = (v4f*)out + b * ((size_t)(CIN + NF) * SEQ / 4) + r;
    *dst = *src;
}

extern "C" void kernel_launch(void* const* d_in, const int* in_sizes, int n_in,
                              void* d_out, int out_size, void* d_ws, size_t ws_size,
                              hipStream_t stream)
{
    const float* x  = (const float*)d_in[0];
    const float* Wq = (const float*)d_in[1];
    const float* bq = (const float*)d_in[2];
    const float* Wk = (const float*)d_in[3];
    const float* bk = (const float*)d_in[4];
    const float* Wv = (const float*)d_in[5];
    const float* bv = (const float*)d_in[6];
    float* out = (float*)d_out;

    // Workspace (~130 MB; scores/probs intermediates stay L2-resident on 192MB L2)
    char* ws = (char*)d_ws;
    const size_t qkvE = (size_t)BATCH * SEQ * NF;
    const size_t ssE  = (size_t)BATCH * SEQ * SEQ;
    char*  qb = ws;                                   // bf16 [B][S][NF]
    char*  kb = qb + qkvE * 2;                        // bf16 [B][S][NF]
    char*  vt = kb + qkvE * 2;                        // bf16 [B][NF][S] (transposed)
    float* sc = (float*)(vt + qkvE * 2);              // f32  [B][S][S]
    char*  pp = (char*)sc + ssE * 4;                  // bf16 [B][S][S]
    float* pm = (float*)(pp + ssE * 2);               // f32  [B][NCH][S]
    float* pl = pm + (size_t)BATCH * NCH * SEQ;       // f32  [B][NCH][S]
    float* cm = pl + (size_t)BATCH * NCH * SEQ;       // f32  [B][S]
    float* ci = cm + (size_t)BATCH * SEQ;             // f32  [B][S]

    dim3 blk(256);
    k_qkv<<<dim3(SEQ / 128, NF / 16, BATCH * 3), blk, 0, stream>>>(
        x, Wq, bq, Wk, bk, Wv, bv, qb, kb, vt);
    k_scores<<<dim3(SEQ / 16, SEQ / 128, BATCH), blk, 0, stream>>>(qb, kb, sc);
    k_sm_part<<<dim3(SEQ / 256, NCH, BATCH), blk, 0, stream>>>(sc, pm, pl);
    k_sm_comb<<<dim3(SEQ / 256, BATCH), blk, 0, stream>>>(pm, pl, cm, ci);
    k_sm_write<<<dim3(SEQ / 256, NCH, BATCH), blk, 0, stream>>>(sc, cm, ci, pp);
    k_pv<<<dim3(NF / 16, SEQ / 128, BATCH), blk, 0, stream>>>(pp, vt, out);
    k_copy_x<<<dim3((unsigned)((size_t)BATCH * CIN * SEQ / 4 / 256)), blk, 0, stream>>>(x, out);

    (void)in_sizes; (void)n_in; (void)out_size; (void)ws_size;
}
